// AttnBlock_61168924229645
// MI455X (gfx1250) — matrix operand order repaired and checked by reading, not compiled, not run on hardware
//
#include <hip/hip_runtime.h>
#include <hip/hip_bf16.h>

typedef __bf16 bf16;
typedef __attribute__((ext_vector_type(16))) __bf16     v16bf;
typedef __attribute__((ext_vector_type(8)))  float      v8f;
typedef __attribute__((ext_vector_type(4)))  unsigned   v4u;

#define ROWS   65536   // B * H * W = 16*4096
#define CDIM   512
#define NSEQ   4096
#define NB     16
#define NH     8
#define DH     64

__device__ __forceinline__ bf16 to_bf16(float f) {
  unsigned u = __builtin_bit_cast(unsigned, f);
  u += 0x7FFFu + ((u >> 16) & 1u);                 // round-to-nearest-even
  unsigned short s = (unsigned short)(u >> 16);
  return __builtin_bit_cast(bf16, s);
}

union Frag16 { v16bf v; v4u q[2]; };

// A operand: row-major [16 x 32] region at p, row stride ld (elements).
// ISA 7.12.2 (16-bit A 16x32): lane m=lane&15; half=lane>>4;
// elems 0..7 -> K = half*8 + 0..7 ; elems 8..15 -> K = 16 + half*8 + 0..7
__device__ __forceinline__ v16bf load_afrag(const bf16* p, int ld) {
  int lane = threadIdx.x & 31;
  const bf16* r = p + (lane & 15) * ld + (lane >> 4) * 8;
  Frag16 f;
  f.q[0] = *(const v4u*)(r);
  f.q[1] = *(const v4u*)(r + 16);
  return f.v;
}

// B operand from an N-major (transposed) [16 x 32] region: row = N, col = K.
// ISA: col n = lane&15; elems 0..7 -> K = half*8 + e ; elems 8..15 -> K = 16 + half*8 + (e-8)
__device__ __forceinline__ v16bf load_bfrag(const bf16* p, int ld) {
  int lane = threadIdx.x & 31;
  const bf16* r = p + (lane & 15) * ld + (lane >> 4) * 8;
  Frag16 f;
  f.q[0] = *(const v4u*)(r);
  f.q[1] = *(const v4u*)(r + 16);
  return f.v;
}

__device__ __forceinline__ v8f wmma_bf16(v16bf a, v16bf b, v8f c) {
  return __builtin_amdgcn_wmma_f32_16x16x32_bf16(false, a, false, b, (short)0, c,
                                                 false, false);
}

// ---- CDNA5 async global->LDS copy (ASYNCcnt-tracked, see 08_async_tensor) --
__device__ __forceinline__ unsigned lds_off_u32(const void* p) {
  // generic -> LDS address space, then to the 32-bit LDS byte offset
  return (unsigned)(unsigned long long)(__attribute__((address_space(3))) const char*)p;
}
__device__ __forceinline__ void async_copy_b128(const void* gsrc, void* ldst) {
  asm volatile("global_load_async_to_lds_b128 %0, %1, off"
               :: "v"(lds_off_u32(ldst)), "v"(gsrc)
               : "memory");
}
__device__ __forceinline__ void wait_async0() {
  asm volatile("s_wait_asynccnt 0x0" ::: "memory");
}

// ---------------- LayerNorm 1 (f32 -> bf16), one wave per 512-wide row -----
__global__ __launch_bounds__(256) void ln1_kernel(const float* __restrict__ x,
                                                  const float* __restrict__ scale,
                                                  bf16* __restrict__ out) {
  int w    = blockIdx.x * 8 + (threadIdx.x >> 5);
  int lane = threadIdx.x & 31;
  const float* xr = x + (size_t)w * CDIM;
  float v[16]; float s = 0.f;
#pragma unroll
  for (int i = 0; i < 16; ++i) { v[i] = xr[lane + i * 32]; s += v[i]; }
#pragma unroll
  for (int o = 16; o >= 1; o >>= 1) s += __shfl_xor(s, o);
  float mean = s * (1.f / 512.f);
  float vs = 0.f;
#pragma unroll
  for (int i = 0; i < 16; ++i) { float d = v[i] - mean; vs += d * d; }
#pragma unroll
  for (int o = 16; o >= 1; o >>= 1) vs += __shfl_xor(vs, o);
  float rstd = rsqrtf(vs * (1.f / 512.f) + 1e-5f);
  bf16* orow = out + (size_t)w * CDIM;
#pragma unroll
  for (int i = 0; i < 16; ++i)
    orow[lane + i * 32] = to_bf16((v[i] - mean) * rstd * scale[lane + i * 32]);
}

// ------------- weight transpose + f32->bf16 : w[K][N] -> wT[N][K] ----------
__global__ __launch_bounds__(256) void transpose_to_bf16(const float* __restrict__ w,
                                                         bf16* __restrict__ wT,
                                                         int K, int N) {
  int i = blockIdx.x * 256 + threadIdx.x;
  if (i >= K * N) return;
  int k = i / N, n = i - k * N;
  wT[(size_t)n * K + k] = to_bf16(w[i]);
}

// ---------------- 128x128 tiled bf16 WMMA GEMM: C = A * Bt^T (+bias) -------
// A: [M x K] row-major (lda=K), Bt: [N x K] row-major (ldb=K), C: [M x N]
// Double-buffered LDS staged with global_load_async_to_lds_b128 (ASYNCcnt).
#define APAD 40
__global__ __launch_bounds__(256) void gemm_bf16_kernel(const bf16* __restrict__ A,
                                                        const bf16* __restrict__ Bt,
                                                        float* __restrict__ C,
                                                        const float* __restrict__ bias,
                                                        int M, int N, int K) {
  __shared__ bf16 As[2][128 * APAD];
  __shared__ bf16 Bs[2][128 * APAD];
  const int t = threadIdx.x, lane = t & 31, wave = t >> 5;
  const int half = lane >> 4, l16 = lane & 15;
  const int m0 = blockIdx.y * 128, n0 = blockIdx.x * 128;
  const int wm = (wave >> 1) * 32, wn = (wave & 1) * 64;

  // two per-thread b128 staging chunks for each of A / B tiles
  const int u0 = t, u1 = t + 256;
  const int row0 = u0 >> 2, c40 = (u0 & 3) * 8;
  const int row1 = u1 >> 2, c41 = (u1 & 3) * 8;
  const bf16* aA0 = &A[(size_t)(m0 + row0) * K + c40];
  const bf16* aA1 = &A[(size_t)(m0 + row1) * K + c41];
  const bf16* aB0 = &Bt[(size_t)(n0 + row0) * K + c40];
  const bf16* aB1 = &Bt[(size_t)(n0 + row1) * K + c41];

  auto stage = [&](int buf, int k0) {
    async_copy_b128(aA0 + k0, &As[buf][row0 * APAD + c40]);
    async_copy_b128(aA1 + k0, &As[buf][row1 * APAD + c41]);
    async_copy_b128(aB0 + k0, &Bs[buf][row0 * APAD + c40]);
    async_copy_b128(aB1 + k0, &Bs[buf][row1 * APAD + c41]);
    if (k0 + 32 < K) {      // WGP-scope prefetch of the tile after next
      __builtin_prefetch(aA0 + k0 + 32, 0, 3);
      __builtin_prefetch(aB0 + k0 + 32, 0, 3);
    }
  };

  const v8f z = {0.f, 0.f, 0.f, 0.f, 0.f, 0.f, 0.f, 0.f};
  v8f acc[2][4];
#pragma unroll
  for (int i = 0; i < 2; ++i)
#pragma unroll
    for (int j = 0; j < 4; ++j) acc[i][j] = z;

  stage(0, 0);
  wait_async0();
  __syncthreads();

  int buf = 0;
  for (int k0 = 0; k0 < K; k0 += 32) {
    const int nxt = buf ^ 1;
    if (k0 + 32 < K) stage(nxt, k0 + 32);   // overlap next tile copy with WMMA

    v16bf af[2], bfg[4];
#pragma unroll
    for (int i = 0; i < 2; ++i)
      af[i] = load_afrag(&As[buf][(wm + i * 16) * APAD], APAD);
#pragma unroll
    for (int j = 0; j < 4; ++j)
      bfg[j] = load_bfrag(&Bs[buf][(wn + j * 16) * APAD], APAD);
#pragma unroll
    for (int i = 0; i < 2; ++i)
#pragma unroll
      for (int j = 0; j < 4; ++j) acc[i][j] = wmma_bf16(af[i], bfg[j], acc[i][j]);

    wait_async0();       // my async writes into nxt have landed in LDS
    __syncthreads();     // everyone done reading buf & writing nxt
    buf = nxt;
  }

#pragma unroll
  for (int i = 0; i < 2; ++i)
#pragma unroll
    for (int j = 0; j < 4; ++j) {
      int row = m0 + wm + i * 16 + half * 8;
      int col = n0 + wn + j * 16 + l16;
      float badd = bias ? bias[col] : 0.f;
#pragma unroll
      for (int r = 0; r < 8; ++r)
        C[(size_t)(row + r) * N + col] = acc[i][j][r] + badd;
    }
}

// ------------- q softmax over d (last 64 of each head) + /8, -> bf16 -------
__global__ __launch_bounds__(256) void qsoftmax_kernel(const float* __restrict__ qkv,
                                                       bf16* __restrict__ qb) {
  size_t w = (size_t)blockIdx.x * 8 + (threadIdx.x >> 5);  // head-row 0..524287
  int lane = threadIdx.x & 31;
  size_t bn = w >> 3; int h = (int)(w & 7);
  const float* p = qkv + bn * 1536 + h * DH;
  float v0 = p[lane], v1 = p[lane + 32];
  float m = fmaxf(v0, v1);
#pragma unroll
  for (int o = 16; o >= 1; o >>= 1) m = fmaxf(m, __shfl_xor(m, o));
  float e0 = __expf(v0 - m), e1 = __expf(v1 - m);
  float s = e0 + e1;
#pragma unroll
  for (int o = 16; o >= 1; o >>= 1) s += __shfl_xor(s, o);
  float inv = 0.125f / s;     // / sqrt(64)
  bf16* q = qb + bn * CDIM + h * DH;
  q[lane]      = to_bf16(e0 * inv);
  q[lane + 32] = to_bf16(e1 * inv);
}

// ------------- k softmax over n (=4096), per (b, channel) column -----------
__global__ __launch_bounds__(256) void ksoftmax_kernel(const float* __restrict__ qkv,
                                                       bf16* __restrict__ kb) {
  int c = blockIdx.x, b = blockIdx.y, t = threadIdx.x;
  const float* base = qkv + (size_t)b * NSEQ * 1536 + 512 + c;
  __shared__ float red[256];
  float v[16]; float m = -3.4e38f;
#pragma unroll
  for (int i = 0; i < 16; ++i) {
    v[i] = base[(size_t)(t + i * 256) * 1536];
    m = fmaxf(m, v[i]);
  }
  red[t] = m; __syncthreads();
  for (int o = 128; o >= 1; o >>= 1) { if (t < o) red[t] = fmaxf(red[t], red[t + o]); __syncthreads(); }
  m = red[0]; __syncthreads();
  float s = 0.f;
#pragma unroll
  for (int i = 0; i < 16; ++i) s += __expf(v[i] - m);
  red[t] = s; __syncthreads();
  for (int o = 128; o >= 1; o >>= 1) { if (t < o) red[t] += red[t + o]; __syncthreads(); }
  float inv = 1.f / red[0];
  bf16* outp = kb + (size_t)b * NSEQ * CDIM + c;
#pragma unroll
  for (int i = 0; i < 16; ++i)
    outp[(size_t)(t + i * 256) * CDIM] = to_bf16(__expf(v[i] - m) * inv);
}

__global__ __launch_bounds__(256) void zero_kernel(float* p, int n) {
  int i = blockIdx.x * 256 + threadIdx.x;
  if (i < n) p[i] = 0.f;
}

// -------- context[b,h,d,e] = sum_n k[n,d] * v[n,e]/n  (K-split + atomics) --
#define CP 40
__global__ __launch_bounds__(256) void context_kernel(const bf16* __restrict__ kb,
                                                      const float* __restrict__ qkv,
                                                      float* __restrict__ ctx) {
  int bh = blockIdx.y, b = bh >> 3, h = bh & 7;
  int nbase = blockIdx.x * 512;
  __shared__ bf16 kT[64 * CP];   // [d][n_local]
  __shared__ bf16 vT[64 * CP];   // [e][n_local]
  int t = threadIdx.x, lane = t & 31, wave = t >> 5;
  int half = lane >> 4, l16 = lane & 15;
  int wm = (wave & 3) * 16, wn = (wave >> 2) * 32;
  const v8f z = {0.f, 0.f, 0.f, 0.f, 0.f, 0.f, 0.f, 0.f};
  v8f acc[2] = {z, z};
  for (int n0 = nbase; n0 < nbase + 512; n0 += 32) {
#pragma unroll
    for (int u = t; u < 2048; u += 256) {
      int d = u & 63, i = u >> 6;
      size_t nrow = (size_t)b * NSEQ + n0 + i;
      kT[d * CP + i] = kb[nrow * CDIM + h * DH + d];
      vT[d * CP + i] = to_bf16(qkv[nrow * 1536 + 1024 + h * DH + d] * (1.f / 4096.f));
    }
    __syncthreads();
    v16bf af = load_afrag(&kT[wm * CP], CP);          // A = k^T  (m=d, k=n)
    acc[0] = wmma_bf16(af, load_bfrag(&vT[wn * CP], CP), acc[0]);
    acc[1] = wmma_bf16(af, load_bfrag(&vT[(wn + 16) * CP], CP), acc[1]);
    __syncthreads();
  }
#pragma unroll
  for (int j = 0; j < 2; ++j)
#pragma unroll
    for (int r = 0; r < 8; ++r) {
      int d = wm + half * 8 + r, e = wn + j * 16 + l16;
      atomicAdd(&ctx[(size_t)bh * 4096 + d * 64 + e], acc[j][r]);
    }
}

// -------- attn[n, h*64+e] = sum_d q[n,h,d] * context[h][d][e]  -> bf16 -----
#define QP 72
__global__ __launch_bounds__(256) void attn_kernel(const bf16* __restrict__ qb,
                                                   const float* __restrict__ ctx,
                                                   bf16* __restrict__ attnb) {
  int bh = blockIdx.y, b = bh >> 3, h = bh & 7;
  int n0 = blockIdx.x * 128;
  __shared__ bf16 Qs[128 * QP];   // [n_local][d]
  __shared__ bf16 Ct[64 * QP];    // [e][d]  (B operand, N-major)
  int t = threadIdx.x, lane = t & 31, wave = t >> 5;
  int half = lane >> 4, l16 = lane & 15;
#pragma unroll
  for (int u = t; u < 4096; u += 256) {
    int e = u & 63, d = u >> 6;
    Ct[e * QP + d] = to_bf16(ctx[(size_t)bh * 4096 + d * 64 + e]);
  }
#pragma unroll
  for (int u = t; u < 1024; u += 256) {
    int row = u >> 3, c = (u & 7) * 8;
    *(v4u*)&Qs[row * QP + c] =
        *(const v4u*)&qb[((size_t)b * NSEQ + n0 + row) * CDIM + h * DH + c];
  }
  __syncthreads();
  const v8f z = {0.f, 0.f, 0.f, 0.f, 0.f, 0.f, 0.f, 0.f};
  v8f acc[4] = {z, z, z, z};
#pragma unroll
  for (int kc = 0; kc < 64; kc += 32) {
    v16bf af = load_afrag(&Qs[(wave * 16) * QP + kc], QP);
#pragma unroll
    for (int j = 0; j < 4; ++j)
      acc[j] = wmma_bf16(af, load_bfrag(&Ct[(j * 16) * QP + kc], QP), acc[j]);
  }
#pragma unroll
  for (int j = 0; j < 4; ++j)
#pragma unroll
    for (int r = 0; r < 8; ++r)
      attnb[((size_t)b * NSEQ + n0 + wave * 16 + half * 8 + r) * CDIM +
            h * DH + j * 16 + l16] = to_bf16(acc[j][r]);
}

// ---------------- LN2 + residual (f32 out) ---------------------------------
__global__ __launch_bounds__(256) void ln2_kernel(const float* __restrict__ proj,
                                                  const float* __restrict__ x,
                                                  const float* __restrict__ scale,
                                                  float* __restrict__ out) {
  int w = blockIdx.x * 8 + (threadIdx.x >> 5);
  int lane = threadIdx.x & 31;
  const float* pr = proj + (size_t)w * CDIM;
  const float* xr = x    + (size_t)w * CDIM;
  float v[16]; float s = 0.f;
#pragma unroll
  for (int i = 0; i < 16; ++i) { v[i] = pr[lane + i * 32]; s += v[i]; }
#pragma unroll
  for (int o = 16; o >= 1; o >>= 1) s += __shfl_xor(s, o);
  float mean = s * (1.f / 512.f);
  float vs = 0.f;
#pragma unroll
  for (int i = 0; i < 16; ++i) { float d = v[i] - mean; vs += d * d; }
#pragma unroll
  for (int o = 16; o >= 1; o >>= 1) vs += __shfl_xor(vs, o);
  float rstd = rsqrtf(vs * (1.f / 512.f) + 1e-5f);
  float* orow = out + (size_t)w * CDIM;
#pragma unroll
  for (int i = 0; i < 16; ++i) {
    int c = lane + i * 32;
    orow[c] = (v[i] - mean) * rstd * scale[c] + xr[c];
  }
}

// ---------------------------------------------------------------------------
extern "C" void kernel_launch(void* const* d_in, const int* in_sizes, int n_in,
                              void* d_out, int out_size, void* d_ws, size_t ws_size,
                              hipStream_t stream) {
  (void)in_sizes; (void)n_in; (void)out_size; (void)ws_size;
  const float* x    = (const float*)d_in[0];
  const float* ln1s = (const float*)d_in[1];
  const float* wqkv = (const float*)d_in[2];
  const float* wout = (const float*)d_in[3];
  const float* bout = (const float*)d_in[4];
  const float* ln2s = (const float*)d_in[5];
  float* out = (float*)d_out;
  char*  ws  = (char*)d_ws;

  const size_t OFF_H   = 0;                       // 64MB  h_bf16 (later attn_bf16)
  const size_t OFF_QKV = OFF_H   + 67108864;      // 384MB qkv f32 (later proj f32)
  const size_t OFF_QB  = OFF_QKV + 402653184;     // 64MB  q bf16
  const size_t OFF_KB  = OFF_QB  + 67108864;      // 64MB  k bf16
  const size_t OFF_WQT = OFF_KB  + 67108864;      // 1.5MB w_qkv^T bf16
  const size_t OFF_WOT = OFF_WQT + 1572864;       // 0.5MB w_out^T bf16
  const size_t OFF_CTX = OFF_WOT + 524288;        // 2MB   context f32

  bf16*  h_b    = (bf16*)(ws + OFF_H);
  float* qkv    = (float*)(ws + OFF_QKV);
  bf16*  q_b    = (bf16*)(ws + OFF_QB);
  bf16*  k_b    = (bf16*)(ws + OFF_KB);
  bf16*  wqT    = (bf16*)(ws + OFF_WQT);
  bf16*  woT    = (bf16*)(ws + OFF_WOT);
  float* ctx    = (float*)(ws + OFF_CTX);
  bf16*  attn_b = h_b;          // alias: h dead after QKV GEMM
  float* proj   = qkv;          // alias: qkv dead after softmaxes + context

  ln1_kernel<<<ROWS / 8, 256, 0, stream>>>(x, ln1s, h_b);
  transpose_to_bf16<<<(512 * 1536) / 256, 256, 0, stream>>>(wqkv, wqT, 512, 1536);
  transpose_to_bf16<<<(512 * 512) / 256, 256, 0, stream>>>(wout, woT, 512, 512);
  gemm_bf16_kernel<<<dim3(1536 / 128, ROWS / 128), 256, 0, stream>>>(
      h_b, wqT, qkv, nullptr, ROWS, 1536, 512);
  qsoftmax_kernel<<<(ROWS * NH) / 8, 256, 0, stream>>>(qkv, q_b);
  ksoftmax_kernel<<<dim3(512, NB), 256, 0, stream>>>(qkv, k_b);
  zero_kernel<<<(NB * NH * 64 * 64) / 256, 256, 0, stream>>>(ctx, NB * NH * 64 * 64);
  context_kernel<<<dim3(8, NB * NH), 256, 0, stream>>>(k_b, qkv, ctx);
  attn_kernel<<<dim3(NSEQ / 128, NB * NH), 256, 0, stream>>>(q_b, ctx, attn_b);
  gemm_bf16_kernel<<<dim3(512 / 128, ROWS / 128), 256, 0, stream>>>(
      attn_b, woT, proj, bout, ROWS, 512, 512);
  ln2_kernel<<<ROWS / 8, 256, 0, stream>>>(proj, x, ln2s, out);
}
